// Conv4Pim_group_arr_v3_10445360464134
// MI455X (gfx1250) — matrix operand order repaired and checked by reading, not compiled, not run on hardware
//
#include <hip/hip_runtime.h>

typedef _Float16 f16_t;
typedef _Float16 v16h __attribute__((ext_vector_type(16)));
typedef _Float16 v8h  __attribute__((ext_vector_type(8)));
typedef float    v8f  __attribute__((ext_vector_type(8)));
typedef unsigned int u32x4 __attribute__((ext_vector_type(4)));
typedef int          i32x4 __attribute__((ext_vector_type(4)));
typedef int          i32x8 __attribute__((ext_vector_type(8)));

#if __has_builtin(__builtin_amdgcn_tensor_load_to_lds)
#define HAVE_TDM 1
#else
#define HAVE_TDM 0
#endif

namespace {

constexpr int kIC    = 256;
constexpr int kOC    = 256;
constexpr int kHdim  = 56;
constexpr int kWdim  = 56;
constexpr int kHW    = kHdim * kWdim;      // 3136
constexpr int kBatch = 16;
constexpr int kGroups = 4;
constexpr int kNumIC  = 64;
constexpr int kKtot   = kNumIC * 9;        // 576
constexpr int kKC     = 32;                // WMMA f16 K
constexpr int kNKC    = kKtot / kKC;       // 18
constexpr int kTileM  = 128;
constexpr int kTileN  = 128;
constexpr int kLdsStr = 40;                // padded f16 row stride (80B): 16B-aligned, bank-conflict-free
constexpr int kNTot   = kBatch * kHW;      // 50176 = 392 * 128

__device__ inline v8f vzero8f() {
  v8f v;
#pragma unroll
  for (int i = 0; i < 8; ++i) v[i] = 0.f;
  return v;
}

__device__ inline v16h ldA(const f16_t* base) {
  v8h lo = *(const v8h*)(base);
  v8h hi = *(const v8h*)(base + 16);
  v16h r;
#pragma unroll
  for (int e = 0; e < 8; ++e) { r[e] = lo[e]; r[e + 8] = hi[e]; }
  return r;
}

__device__ inline v16h ldB(const f16_t* base) {
  v8h lo = *(const v8h*)(base);
  v8h hi = *(const v8h*)(base + 16);
  v16h r;
#pragma unroll
  for (int e = 0; e < 8; ++e) { r[e] = lo[e]; r[e + 8] = hi[e]; }
  return r;
}

#if HAVE_TDM
// TDM: load a 128(row) x 32(f16) tile, row stride 576 f16, into LDS with a
// 16B pad after every 64B row (LDS row stride 80B == kLdsStr f16).
__device__ inline void tdm_load_tile(unsigned lds_byte_off, unsigned long long gaddr) {
  u32x4 g0;
  g0[0] = 1u;                                       // count=1, user desc
  g0[1] = lds_byte_off;                             // lds_addr
  g0[2] = (unsigned)gaddr;                          // global_addr[31:0]
  g0[3] = ((unsigned)(gaddr >> 32) & 0x01FFFFFFu)   // global_addr[56:32]
        | (2u << 30);                               // type=2 (image)
  i32x8 g1;
  g1[0] = (int)((1u << 16)    // data_size = 2 bytes
              | (1u << 20)    // pad_enable
              | (3u << 22)    // pad_interval: 16 DWORDs (one 64B tile row)
              | (3u << 25));  // pad_amount: 4 DWORDs (16B)
  g1[1] = (int)(576u << 16);  // tensor_dim0[15:0] at bits[63:48]
  g1[2] = (int)(256u << 16);  // tensor_dim0[31:16]=0 | tensor_dim1[15:0]=256
  g1[3] = (int)(32u << 16);   // tensor_dim1[31:16]=0 | tile_dim0=32
  g1[4] = 128;                // tile_dim1=128 | tile_dim2=0
  g1[5] = 576;                // tensor_dim0_stride[31:0]
  g1[6] = 0;                  // stride0[47:32] | tensor_dim1_stride[15:0]
  g1[7] = 0;
  i32x4 z4 = {0, 0, 0, 0};
#if __has_include(<hip/amd_detail/amd_gfx1250_TDM.h>)
  i32x8 z8 = {0, 0, 0, 0, 0, 0, 0, 0};
  __builtin_amdgcn_tensor_load_to_lds(g0, g1, z4, z4, z8, 0);
#else
  __builtin_amdgcn_tensor_load_to_lds(g0, g1, z4, z4, 0);
#endif
}
#endif

__device__ inline unsigned lds_off_of(const void* p) {
  // LDS aperture: flat addr low 32 bits carry the LDS byte offset.
  return (unsigned)(unsigned long long)(uintptr_t)p;
}

// ---------------- Pre-pass: LSQ-quantize weights to f16 integers -------------
// Layout: wq[((g*2 + sign)*256 + oc)*576 + (ic_local*9 + kh*3 + kw)]
__global__ __launch_bounds__(256)
void quant_weights_kernel(const float* __restrict__ weight,
                          const float* __restrict__ w_scale,
                          f16_t* __restrict__ wq)
{
  int idx = blockIdx.x * 256 + threadIdx.x;     // < 256*256*9 = 589824
  int oc  = idx / 2304;                         // 2304 = 256*9
  int rem = idx - oc * 2304;
  int ic  = rem / 9;
  int r   = rem - ic * 9;
  int g   = ic >> 6;
  int k   = (ic & 63) * 9 + r;
  float w = weight[idx];
  float s = w_scale[g];
  float qp = rintf(fminf(fmaxf( w / s, 0.f), 3.f));
  float qn = rintf(fminf(fmaxf(-w / s, 0.f), 3.f));
  wq[((size_t)(g * 2 + 0) * 256 + oc) * 576 + k] = (f16_t)qp;
  wq[((size_t)(g * 2 + 1) * 256 + oc) * 576 + k] = (f16_t)qn;
}

// ---------------- Main implicit-GEMM WMMA kernel -----------------------------
__global__ __launch_bounds__(256)
void conv4pim_wmma_kernel(const float* __restrict__ x,
                          const f16_t* __restrict__ wq,
                          const float* __restrict__ w_scale,
                          const float* __restrict__ ps_scale_p,
                          const float* __restrict__ ps_scale_n,
                          float* __restrict__ out)
{
  __shared__ f16_t sAp[kTileM * kLdsStr];   // quantized +weights, [M][K] padded
  __shared__ f16_t sAn[kTileM * kLdsStr];   // quantized -weights
  __shared__ f16_t sB [kTileN * kLdsStr];   // im2col x tile, [N][K] padded

  const int m0   = blockIdx.y * kTileM;
  const int n0   = blockIdx.x * kTileN;
  const int tid  = threadIdx.x;
  const int lane = tid & 31;
  const int wave = tid >> 5;
  const int wm   = (wave & 1) * 64;
  const int wn   = (wave >> 1) * 32;

  // B staging: 2 threads per n-row, 16 k each
  const int bn    = tid >> 1;
  const int bkoff = (tid & 1) * 16;
  const int n_g   = n0 + bn;
  const int b_img = n_g / kHW;
  const int hw    = n_g - b_img * kHW;
  const int oh    = hw / kWdim;
  const int ow    = hw - oh * kWdim;
  const float* xb = x + (size_t)b_img * kIC * kHW;

#if !HAVE_TDM
  const int ar    = tid & 127;
  const int ahalf = (tid >> 7) * 16;
#endif

  const int rsel = lane & 15;
  const int hi   = lane >> 4;

  const unsigned offAp = lds_off_of(&sAp[0]);
  const unsigned offAn = lds_off_of(&sAn[0]);
  const unsigned long long wqbase = (unsigned long long)(uintptr_t)wq;

  v8f accP[4][2], accN[4][2], outAcc[4][2];
#pragma unroll
  for (int i = 0; i < 4; ++i)
#pragma unroll
    for (int j = 0; j < 2; ++j) outAcc[i][j] = vzero8f();

  for (int g = 0; g < kGroups; ++g) {
    const float sw = w_scale[g];
    const float sp = ps_scale_p[g];
    const float sn = ps_scale_n[g];
    const float ap = sw / sp;
    const float an = sw / sn;

#pragma unroll
    for (int i = 0; i < 4; ++i)
#pragma unroll
      for (int j = 0; j < 2; ++j) { accP[i][j] = vzero8f(); accN[i][j] = vzero8f(); }

    const float* xg = xb + (size_t)g * kNumIC * kHW;

    for (int kc = 0; kc < kNKC; ++kc) {
      const int k0 = kc * kKC;
      __syncthreads();   // all waves done reading previous tiles

#if HAVE_TDM
      if (wave == 0) {
        unsigned long long gaP =
            wqbase + (((size_t)(g * 2 + 0) * 256 + m0) * 576 + k0) * 2ull;
        unsigned long long gaN =
            wqbase + (((size_t)(g * 2 + 1) * 256 + m0) * 576 + k0) * 2ull;
        tdm_load_tile(offAp, gaP);
        tdm_load_tile(offAn, gaN);
      }
#else
      {
        const f16_t* srcP = wq + ((size_t)(g * 2 + 0) * 256 + m0 + ar) * 576 + k0 + ahalf;
        const f16_t* srcN = wq + ((size_t)(g * 2 + 1) * 256 + m0 + ar) * 576 + k0 + ahalf;
        *(uint4*)&sAp[ar * kLdsStr + ahalf]     = *(const uint4*)(srcP);
        *(uint4*)&sAp[ar * kLdsStr + ahalf + 8] = *(const uint4*)(srcP + 8);
        *(uint4*)&sAn[ar * kLdsStr + ahalf]     = *(const uint4*)(srcN);
        *(uint4*)&sAn[ar * kLdsStr + ahalf + 8] = *(const uint4*)(srcN + 8);
      }
#endif

      // ---- stage B: im2col of x, f32 -> f16 ----
      {
        f16_t* dst = &sB[bn * kLdsStr + bkoff];
#pragma unroll
        for (int j = 0; j < 16; ++j) {
          int k  = k0 + bkoff + j;
          int ic = k / 9;
          int r  = k - ic * 9;
          int kh = r / 3;
          int kw = r - kh * 3;
          int ih = oh + kh - 1;
          int iw = ow + kw - 1;
          float v = 0.f;
          if ((unsigned)ih < (unsigned)kHdim && (unsigned)iw < (unsigned)kWdim)
            v = xg[ic * kHW + ih * kWdim + iw];
          dst[j] = (f16_t)v;
        }
      }

#if HAVE_TDM
      if (wave == 0) __builtin_amdgcn_s_wait_tensorcnt(0);
#endif
      __syncthreads();

      // ---- fragments + WMMA ----
      v16h bfrag[2];
#pragma unroll
      for (int j = 0; j < 2; ++j)
        bfrag[j] = ldB(&sB[(wn + j * 16 + rsel) * kLdsStr + hi * 8]);

#pragma unroll
      for (int i = 0; i < 4; ++i) {
        v16h aP = ldA(&sAp[(wm + i * 16 + rsel) * kLdsStr + hi * 8]);
        v16h aN = ldA(&sAn[(wm + i * 16 + rsel) * kLdsStr + hi * 8]);
#pragma unroll
        for (int j = 0; j < 2; ++j) {
          accP[i][j] = __builtin_amdgcn_wmma_f32_16x16x32_f16(
              false, aP, false, bfrag[j], (short)0, accP[i][j], false, false);
          accN[i][j] = __builtin_amdgcn_wmma_f32_16x16x32_f16(
              false, aN, false, bfrag[j], (short)0, accN[i][j], false, false);
        }
      }
    }

    // ---- per-group epilogue: LSQ-quantize partial sums, fold p - n ----
#pragma unroll
    for (int i = 0; i < 4; ++i)
#pragma unroll
      for (int j = 0; j < 2; ++j)
#pragma unroll
        for (int e = 0; e < 8; ++e) {
          float pp = rintf(fminf(fmaxf(accP[i][j][e] * ap, -128.f), 127.f)) * sp;
          float nn = rintf(fminf(fmaxf(accN[i][j][e] * an, -128.f), 127.f)) * sn;
          outAcc[i][j][e] += pp - nn;
        }
  }

  // ---- store: C layout M = e + 8*hi, N = lane&15 ----
  const int hi8 = hi * 8;
#pragma unroll
  for (int i = 0; i < 4; ++i)
#pragma unroll
    for (int j = 0; j < 2; ++j)
#pragma unroll
      for (int e = 0; e < 8; ++e) {
        int m   = m0 + wm + i * 16 + e + hi8;
        int n   = n0 + wn + j * 16 + rsel;
        int b   = n / kHW;
        int hw2 = n - b * kHW;
        out[(size_t)b * (kOC * kHW) + (size_t)m * kHW + hw2] = outAcc[i][j][e];
      }
}

} // namespace

extern "C" void kernel_launch(void* const* d_in, const int* in_sizes, int n_in,
                              void* d_out, int out_size, void* d_ws, size_t ws_size,
                              hipStream_t stream) {
  (void)in_sizes; (void)n_in; (void)out_size; (void)ws_size;
  const float* x       = (const float*)d_in[0];
  const float* weight  = (const float*)d_in[1];
  const float* w_scale = (const float*)d_in[2];
  const float* ps_p    = (const float*)d_in[3];
  const float* ps_n    = (const float*)d_in[4];
  float* out           = (float*)d_out;
  f16_t* wq            = (f16_t*)d_ws;   // 4*2*256*576 f16 = 2.36 MB

  quant_weights_kernel<<<dim3(2304), dim3(256), 0, stream>>>(weight, w_scale, wq);

  dim3 grid(kNTot / kTileN, kOC / kTileM, 1);  // (392, 2)
  conv4pim_wmma_kernel<<<grid, dim3(256), 0, stream>>>(x, wq, w_scale, ps_p, ps_n, out);
}
